// VRNNSimple_18708877541957
// MI455X (gfx1250) — matrix operand order repaired and checked by reading, not compiled, not run on hardware
//
#include <hip/hip_runtime.h>
#include <hip/hip_bf16.h>

typedef __bf16 bf16;
typedef __attribute__((ext_vector_type(16))) __bf16 bf16x16;
typedef __attribute__((ext_vector_type(8)))  __bf16 bf16x8;
typedef __attribute__((ext_vector_type(8)))  float   f32x8;

#define INPUT   64
#define HIDDEN  256
#define ZDIM    32
#define BATCH   256
#define TSTEPS  1024
#define MB      16          // batch rows per block
#define NTHREADS 512        // 16 waves (wave32)

// d_ws byte offsets for bf16 weight copies
#define WS_ENCW  0                       // [64][320]  bf16 = 40960 B (mu rows 0-31, sigma rows 32-63)
#define WS_WIH   40960                   // [768][96]  bf16 = 147456 B
#define WS_WHH   (40960 + 147456)        // [768][256] bf16 = 393216 B

// LDS layout (bytes)
#define OFF_H     0        // [16][256] f32 state            16384
#define OFF_GH    16384    // [16][768] f32 gh, then gh+gi   49152
#define OFF_GIN   65536    // [16][256] f32 i_n (n-gate gi)  16384
#define OFF_ZOUT  81920    // [16][64]  f32 mu|logvar         4096
#define OFF_ENCB  86016    // [64] f32                         256
#define OFF_BIH   86272    // [768] f32                       3072
#define OFF_BHH   89344    // [768] f32                       3072
#define OFF_XH    92416    // [16][320] bf16 x_t|h           10240
#define OFF_XZ    102656   // [16][96]  bf16 x_t|z            3072
#define OFF_ENCW  105728   // [64][320] bf16                 40960
#define OFF_WIH   146688   // [768][96] bf16                147456
#define SMEM_BYTES 294144  // 287 KB of the 320 KB WGP LDS

// ---- one-time weight conversion fp32 -> bf16 into d_ws ----
__global__ void vrnn_cvt_weights(const float* __restrict__ emuW, const float* __restrict__ esigW,
                                 const float* __restrict__ wih,  const float* __restrict__ whh,
                                 bf16* __restrict__ encW, bf16* __restrict__ Wih, bf16* __restrict__ Whh)
{
    int i = blockIdx.x * blockDim.x + threadIdx.x;
    if (i < 64 * 320) {
        int r = i / 320, c = i % 320;
        float v = (r < 32) ? emuW[r * 320 + c] : esigW[(r - 32) * 320 + c];
        encW[i] = (bf16)v;
    }
    if (i < 768 * 96)  Wih[i] = (bf16)wih[i];
    if (i < 768 * 256) Whh[i] = (bf16)whh[i];
}

// A-fragment (16xK row-major bf16 in LDS): lane L holds row M=L&15,
// K in [k+16*half, +16) (elems 0-15)
__device__ __forceinline__ bf16x16 frag_a(const bf16* base, int stride, int M, int half, int k)
{
    const bf16x8* p0 = reinterpret_cast<const bf16x8*>(base + M * stride + k + 16 * half);
    const bf16x8* p1 = reinterpret_cast<const bf16x8*>(base + M * stride + k + 16 * half + 8);
    bf16x8 lo = *p0, hi = *p1;
    return __builtin_shufflevector(lo, hi, 0,1,2,3,4,5,6,7,8,9,10,11,12,13,14,15);
}

// B-fragment: B[k][n] = W[n][k], row-major W[N][K] (global or LDS bf16).
// lane L holds column n = n0 + (L&15), K in [k + 16*half, +16) -> 32 contiguous bytes of row n.
__device__ __forceinline__ bf16x16 frag_b(const bf16* W, int K, int n, int half, int k)
{
    return *reinterpret_cast<const bf16x16*>(W + (size_t)n * K + k + 16 * half);
}

__global__ void __launch_bounds__(NTHREADS)
vrnn_kernel(const float* __restrict__ x, const float* __restrict__ eps,
            const float* __restrict__ emu_b, const float* __restrict__ esig_b,
            const float* __restrict__ bih_g, const float* __restrict__ bhh_g,
            const bf16* __restrict__ encW_g, const bf16* __restrict__ Wih_g,
            const bf16* __restrict__ Whh, float* __restrict__ out)
{
    extern __shared__ char smem[];
    float* h    = (float*)(smem + OFF_H);
    float* gh   = (float*)(smem + OFF_GH);
    float* gin  = (float*)(smem + OFF_GIN);
    float* zout = (float*)(smem + OFF_ZOUT);
    float* encb = (float*)(smem + OFF_ENCB);
    float* bih  = (float*)(smem + OFF_BIH);
    float* bhh  = (float*)(smem + OFF_BHH);
    bf16*  xh   = (bf16*)(smem + OFF_XH);
    bf16*  xz   = (bf16*)(smem + OFF_XZ);
    bf16*  encW = (bf16*)(smem + OFF_ENCW);
    bf16*  Wih  = (bf16*)(smem + OFF_WIH);

    const int tid  = threadIdx.x;
    const int wave = tid >> 5;
    const int lane = tid & 31;
    const int nl   = lane & 15;     // N (or M) coordinate within tile
    const int half = lane >> 4;
    const int b0   = blockIdx.x * MB;

    // ---- init: state, biases, LDS-resident weights ----
    for (int i = tid; i < MB * HIDDEN; i += NTHREADS) h[i] = 0.f;
    for (int i = tid; i < 64; i += NTHREADS) encb[i] = (i < 32) ? emu_b[i] : esig_b[i - 32];
    for (int i = tid; i < 768; i += NTHREADS) { bih[i] = bih_g[i]; bhh[i] = bhh_g[i]; }
    {   // copy bf16 weights into LDS as dwords
        const unsigned* src0 = (const unsigned*)encW_g;
        unsigned*       dst0 = (unsigned*)encW;
        for (int i = tid; i < 64 * 320 / 2; i += NTHREADS) dst0[i] = src0[i];
        const unsigned* src1 = (const unsigned*)Wih_g;
        unsigned*       dst1 = (unsigned*)Wih;
        for (int i = tid; i < 768 * 96 / 2; i += NTHREADS) dst1[i] = src1[i];
    }
    // stage x_0 and bf16(h)=0 into activation buffers
    for (int e = tid; e < MB * INPUT; e += NTHREADS) {
        int row = e >> 6, c = e & 63;
        bf16 bv = (bf16)x[((size_t)(b0 + row) * TSTEPS) * INPUT + c];
        xh[row * 320 + c] = bv;
        xz[row * 96 + c]  = bv;
    }
    for (int e = tid; e < MB * HIDDEN; e += NTHREADS) {
        int row = e >> 8, j = e & 255;
        xh[row * 320 + 64 + j] = (bf16)0.f;
    }

    // ---- persistent W_hh B-fragments: loaded ONCE, reused for all 1024 steps ----
    // 3 tiles x 8 k-steps x 8 VGPRs = 192 VGPRs per wave (block total == whole W_hh)
    bf16x16 whh_f[3][8];
    #pragma unroll
    for (int ti = 0; ti < 3; ++ti) {
        int n0 = (wave * 3 + ti) * 16;
        #pragma unroll
        for (int kk = 0; kk < 8; ++kk)
            whh_f[ti][kk] = frag_b(Whh, 256, n0 + nl, half, kk * 32);
    }
    __syncthreads();

    const int zrow = tid >> 5, zc = tid & 31;   // this thread's (row, z-col), 512 == 16x32

    for (int t = 0; t < TSTEPS; ++t) {
        // issue eps_t load early: latency overlaps the whole GEMM phase
        float ev = eps[((size_t)t * BATCH + b0 + zrow) * ZDIM + zc];

        // ---- P1a: encoder GEMMs (waves 0..3, one 16-wide tile each, K=320, LDS weights) ----
        if (wave < 4) {
            float bv = encb[wave * 16 + nl];
            f32x8 acc;
            #pragma unroll
            for (int r = 0; r < 8; ++r) acc[r] = bv;
            #pragma unroll
            for (int k = 0; k < 320; k += 32) {
                bf16x16 a = frag_a(xh, 320, nl, half, k);
                bf16x16 b = frag_b(encW, 320, wave * 16 + nl, half, k);
                acc = __builtin_amdgcn_wmma_f32_16x16x32_bf16(false, a, false, b,
                                                              (short)0, acc, false, false);
            }
            #pragma unroll
            for (int r = 0; r < 8; r++)
                zout[(r + 8 * half) * 64 + wave * 16 + nl] = acc[r];
        }
        // ---- P1b: gh = h @ W_hh^T + b_hh  (all 16 waves, 3 tiles each; weights in VGPRs) ----
        {
            f32x8 acc[3];
            #pragma unroll
            for (int ti = 0; ti < 3; ++ti) {
                float bv = bhh[wave * 48 + ti * 16 + nl];
                #pragma unroll
                for (int r = 0; r < 8; ++r) acc[ti][r] = bv;
            }
            #pragma unroll
            for (int kk = 0; kk < 8; ++kk) {
                bf16x16 a = frag_a(xh, 320, nl, half, 64 + kk * 32);   // shared by all 3 tiles
                #pragma unroll
                for (int ti = 0; ti < 3; ++ti)
                    acc[ti] = __builtin_amdgcn_wmma_f32_16x16x32_bf16(false, a, false,
                                whh_f[ti][kk], (short)0, acc[ti], false, false);
            }
            #pragma unroll
            for (int ti = 0; ti < 3; ++ti)
                #pragma unroll
                for (int r = 0; r < 8; r++)
                    gh[(r + 8 * half) * 768 + wave * 48 + ti * 16 + nl] = acc[ti][r];
        }
        __syncthreads();

        // ---- P2: reparameterize z = mu + eps * exp(0.5*logvar) -> xz bf16 ----
        {
            float mu = zout[zrow * 64 + zc];
            float lv = zout[zrow * 64 + 32 + zc];
            float z  = mu + ev * __expf(0.5f * lv);
            xz[zrow * 96 + 64 + zc] = (bf16)z;
        }
        __syncthreads();

        // ---- P3: gi = [x_t, z] @ W_ih^T + b_ih (LDS weights);
        //      r/u tiles merged in-place into gh, n tiles into gin ----
        {
            f32x8 acc[3];
            #pragma unroll
            for (int ti = 0; ti < 3; ++ti) {
                float bv = bih[wave * 48 + ti * 16 + nl];
                #pragma unroll
                for (int r = 0; r < 8; ++r) acc[ti][r] = bv;
            }
            #pragma unroll
            for (int kk = 0; kk < 3; ++kk) {
                bf16x16 a = frag_a(xz, 96, nl, half, kk * 32);
                #pragma unroll
                for (int ti = 0; ti < 3; ++ti) {
                    bf16x16 b = frag_b(Wih, 96, wave * 48 + ti * 16 + nl, half, kk * 32);
                    acc[ti] = __builtin_amdgcn_wmma_f32_16x16x32_bf16(false, a, false, b,
                                (short)0, acc[ti], false, false);
                }
            }
            #pragma unroll
            for (int ti = 0; ti < 3; ++ti) {
                int n0 = wave * 48 + ti * 16;       // wave-uniform
                if (n0 < 512) {                      // r/u gates: gh += gi (own columns)
                    #pragma unroll
                    for (int r = 0; r < 8; r++) {
                        int idx = (r + 8 * half) * 768 + n0 + nl;
                        gh[idx] += acc[ti][r];
                    }
                } else {                             // n gate: keep i_n separate
                    #pragma unroll
                    for (int r = 0; r < 8; r++)
                        gin[(r + 8 * half) * 256 + (n0 - 512) + nl] = acc[ti][r];
                }
            }
        }
        __syncthreads();

        // ---- P4: GRU gate math (fp32), update h, write bf16 h, stage x_{t+1} ----
        for (int e = tid; e < MB * HIDDEN; e += NTHREADS) {
            int row = e >> 8, j = e & 255;
            float rg  = gh[row * 768 + j];              // gi+gh merged
            float ug  = gh[row * 768 + 256 + j];        // gi+gh merged
            float in_ = gin[row * 256 + j];             // i_n
            float hn  = gh[row * 768 + 512 + j];        // h_n
            float r = 1.f / (1.f + __expf(-rg));
            float u = 1.f / (1.f + __expf(-ug));
            float n = tanhf(in_ + r * hn);
            float hv = h[row * HIDDEN + j];
            float hnew = (1.f - u) * n + u * hv;
            h[row * HIDDEN + j] = hnew;
            xh[row * 320 + 64 + j] = (bf16)hnew;        // bf16 copy for next step's GEMMs
        }
        if (t + 1 < TSTEPS) {
            for (int e = tid; e < MB * INPUT; e += NTHREADS) {
                int row = e >> 6, c = e & 63;
                const float* xp = &x[((size_t)(b0 + row) * TSTEPS + (t + 1)) * INPUT + c];
                float v = *xp;
                if (t + 2 < TSTEPS) __builtin_prefetch(xp + INPUT, 0, 1);
                bf16 bv = (bf16)v;
                xh[row * 320 + c] = bv;
                xz[row * 96 + c]  = bv;
            }
        }
        __syncthreads();
    }

    // ---- epilogue: h_last [256][256] ----
    for (int e = tid; e < MB * HIDDEN; e += NTHREADS) {
        int row = e >> 8, j = e & 255;
        out[(size_t)(b0 + row) * HIDDEN + j] = h[row * HIDDEN + j];
    }
}

extern "C" void kernel_launch(void* const* d_in, const int* in_sizes, int n_in,
                              void* d_out, int out_size, void* d_ws, size_t ws_size,
                              hipStream_t stream)
{
    const float* x     = (const float*)d_in[0];
    const float* eps   = (const float*)d_in[1];
    const float* emuW  = (const float*)d_in[2];
    const float* emub  = (const float*)d_in[3];
    const float* esigW = (const float*)d_in[4];
    const float* esigb = (const float*)d_in[5];
    // d_in[6..9] = prior weights/biases: dead code in reference (DCE'd)
    const float* wih   = (const float*)d_in[10];
    const float* whh   = (const float*)d_in[11];
    const float* bih   = (const float*)d_in[12];
    const float* bhh   = (const float*)d_in[13];

    bf16* encW = (bf16*)((char*)d_ws + WS_ENCW);
    bf16* Wih  = (bf16*)((char*)d_ws + WS_WIH);
    bf16* Whh  = (bf16*)((char*)d_ws + WS_WHH);

    vrnn_cvt_weights<<<(768 * 256 + 255) / 256, 256, 0, stream>>>(emuW, esigW, wih, whh,
                                                                  encW, Wih, Whh);

    vrnn_kernel<<<BATCH / MB, NTHREADS, SMEM_BYTES, stream>>>(
        x, eps, emub, esigb, bih, bhh, encW, Wih, Whh, (float*)d_out);
}
